// MLP_66855460930188
// MI455X (gfx1250) — compile-verified, matrix operand order checked by reading, not run on hardware
//
#include <hip/hip_runtime.h>
#include <stdint.h>

typedef __attribute__((ext_vector_type(16))) __bf16   v16bf;
typedef __attribute__((ext_vector_type(8)))  float    v8f;
typedef __attribute__((ext_vector_type(4)))  uint32_t u32x4;

union FragBF {
    v16bf v;
    u32x4 q[2];
};

__device__ __forceinline__ uint16_t f2bf(float f) {
    uint32_t u = __float_as_uint(f);
    uint32_t r = u + 0x7FFFu + ((u >> 16) & 1u);   // round-to-nearest-even
    return (uint16_t)(r >> 16);
}

// ---- weight f32 -> bf16 pre-pass (W1: 512x256, W2: 256x512) ----
__global__ void cvt_weights_kernel(const float* __restrict__ W1,
                                   const float* __restrict__ W2,
                                   uint16_t* __restrict__ W1b,
                                   uint16_t* __restrict__ W2b) {
    int i = blockIdx.x * blockDim.x + threadIdx.x;
    if (i < 512 * 256) {
        W1b[i] = f2bf(W1[i]);
        W2b[i] = f2bf(W2[i]);
    }
}

// LDS layout (bytes); padded row strides kill 16-lane bank conflicts
#define X_STRIDE   264              // bf16 elems per row (256 + 8 pad)
#define H1_STRIDE  520              // bf16 elems per row (512 + 8 pad)
#define H2_STRIDE  260              // f32  elems per row (256 + 4 pad)
#define X_OFF      0
#define H1_OFF     (64 * X_STRIDE * 2)                 // 33792
#define H2_OFF     (H1_OFF + 64 * H1_STRIDE * 2)       // 100352
#define SMEM_BYTES (H2_OFF + 64 * H2_STRIDE * 4)       // 166912

__global__ __launch_bounds__(256, 2)
void mlp_edge_kernel(const float* __restrict__ nf,
                     const long long* __restrict__ eli,
                     const uint16_t* __restrict__ W1b,
                     const uint16_t* __restrict__ W2b,
                     const float* __restrict__ b1,
                     const float* __restrict__ b2,
                     const float* __restrict__ W3,
                     const float* __restrict__ b3,
                     float* __restrict__ out,
                     int E) {
    extern __shared__ char smem[];
    uint16_t* Xs  = (uint16_t*)(smem + X_OFF);
    uint16_t* H1s = (uint16_t*)(smem + H1_OFF);
    float*    H2s = (float*)   (smem + H2_OFF);

    const int tid  = threadIdx.x;
    const int wave = tid >> 5;
    const int lane = tid & 31;
    const int lrow = lane & 15;     // A: local m ; B/D: local n
    const int khalf = lane >> 4;    // K half selector per WMMA layout
    const long long base = (long long)blockIdx.x * 64;

    // ---- gather + f32->bf16 into X tile [64 x 256] ----
    {
        int e = tid >> 2;           // 0..63
        int qd = tid & 3;           // quarter: 64 floats each
        long long eg = base + e;
        uint16_t* xrow = Xs + e * X_STRIDE + qd * 64;
        if (eg < (long long)E) {
            long long node = (qd < 2) ? eli[eg] : eli[(long long)E + eg];
            const float4* src = (const float4*)(nf + node * 128 + (qd & 1) * 64);
#pragma unroll
            for (int j = 0; j < 16; ++j) {
                float4 f = src[j];
                uint32_t lo = (uint32_t)f2bf(f.x) | ((uint32_t)f2bf(f.y) << 16);
                uint32_t hi = (uint32_t)f2bf(f.z) | ((uint32_t)f2bf(f.w) << 16);
                ((uint2*)xrow)[j] = make_uint2(lo, hi);
            }
        } else {
#pragma unroll
            for (int j = 0; j < 16; ++j)
                ((uint2*)xrow)[j] = make_uint2(0u, 0u);
        }
    }
    __syncthreads();

    // ---- layer 1: [64x256] x [256->512]  (bf16 WMMA, f32 acc) ----
    for (int t = wave; t < 128; t += 8) {
        int mt = t & 3;
        int nt = t >> 2;            // 0..31
        v8f acc = {};
        const uint16_t* abase = Xs  + (mt * 16 + lrow) * X_STRIDE;
        const uint16_t* bbase = W1b + (nt * 16 + lrow) * 256;
#pragma unroll
        for (int kc = 0; kc < 256; kc += 32) {
            FragBF a, b;
            a.q[0] = *(const u32x4*)(abase + kc + 8 * khalf);
            a.q[1] = *(const u32x4*)(abase + kc + 16 + 8 * khalf);
            b.q[0] = *(const u32x4*)(bbase + kc + 8 * khalf);
            b.q[1] = *(const u32x4*)(bbase + kc + 16 + 8 * khalf);
            acc = __builtin_amdgcn_wmma_f32_16x16x32_bf16(
                false, a.v, false, b.v, (short)0, acc, false, false);
        }
        float bias = b1[nt * 16 + lrow];
        uint16_t* h1p = H1s + (mt * 16 + khalf * 8) * H1_STRIDE + nt * 16 + lrow;
#pragma unroll
        for (int v = 0; v < 8; ++v) {
            float r = fmaxf(acc[v] + bias, 0.0f);
            h1p[v * H1_STRIDE] = f2bf(r);
        }
    }
    __syncthreads();

    // ---- layer 2: [64x512] x [512->256] ----
    for (int t = wave; t < 64; t += 8) {
        int mt = t & 3;
        int nt = t >> 2;            // 0..15
        v8f acc = {};
        const uint16_t* abase = H1s + (mt * 16 + lrow) * H1_STRIDE;
        const uint16_t* bbase = W2b + (nt * 16 + lrow) * 512;
#pragma unroll
        for (int kc = 0; kc < 512; kc += 32) {
            FragBF a, b;
            a.q[0] = *(const u32x4*)(abase + kc + 8 * khalf);
            a.q[1] = *(const u32x4*)(abase + kc + 16 + 8 * khalf);
            b.q[0] = *(const u32x4*)(bbase + kc + 8 * khalf);
            b.q[1] = *(const u32x4*)(bbase + kc + 16 + 8 * khalf);
            acc = __builtin_amdgcn_wmma_f32_16x16x32_bf16(
                false, a.v, false, b.v, (short)0, acc, false, false);
        }
        float bias = b2[nt * 16 + lrow];
        float* h2p = H2s + (mt * 16 + khalf * 8) * H2_STRIDE + nt * 16 + lrow;
#pragma unroll
        for (int v = 0; v < 8; ++v)
            h2p[v * H2_STRIDE] = fmaxf(acc[v] + bias, 0.0f);
    }
    __syncthreads();

    // ---- layer 3: logits = h2 . W3 + b3 (f32, 4 lanes per edge) ----
    {
        int e = tid >> 2;
        int qd = tid & 3;
        const float* hp = H2s + e * H2_STRIDE + qd * 64;
        const float* w3 = W3 + qd * 64;
        float s = 0.0f;
#pragma unroll
        for (int j = 0; j < 64; ++j) s += hp[j] * w3[j];
        s += __shfl_xor(s, 1, 32);
        s += __shfl_xor(s, 2, 32);
        long long eg = base + e;
        if (qd == 0 && eg < (long long)E) out[eg] = s + b3[0];
    }

    // ---- coalesced store of intermediate h2 [E x 256] ----
    {
        float* outh = out + E;
        for (int idx = tid; idx < 64 * 256; idx += 256) {
            int m = idx >> 8;
            int n = idx & 255;
            long long eg = base + m;
            if (eg < (long long)E) outh[eg * 256 + n] = H2s[m * H2_STRIDE + n];
        }
    }
}

extern "C" void kernel_launch(void* const* d_in, const int* in_sizes, int n_in,
                              void* d_out, int out_size, void* d_ws, size_t ws_size,
                              hipStream_t stream) {
    const float*     nf  = (const float*)d_in[0];
    const long long* eli = (const long long*)d_in[1];
    const float*     W1  = (const float*)d_in[2];
    const float*     b1  = (const float*)d_in[3];
    const float*     W2  = (const float*)d_in[4];
    const float*     b2  = (const float*)d_in[5];
    const float*     W3  = (const float*)d_in[6];
    const float*     b3  = (const float*)d_in[7];
    float* out = (float*)d_out;

    uint16_t* W1b = (uint16_t*)d_ws;
    uint16_t* W2b = W1b + 512 * 256;

    const int E = in_sizes[1] / 2;   // 500000

    cvt_weights_kernel<<<(512 * 256 + 255) / 256, 256, 0, stream>>>(W1, W2, W1b, W2b);

    int nblk = (E + 63) / 64;
    mlp_edge_kernel<<<nblk, 256, SMEM_BYTES, stream>>>(
        nf, eli, W1b, W2b, b1, b2, W3, b3, out, E);
}
